// PhysicsInformedEncoder_62843961475163
// MI455X (gfx1250) — matrix operand order repaired and checked by reading, not compiled, not run on hardware
//
#include <hip/hip_runtime.h>
#include <hip/hip_bf16.h>

#define N_NODES 10000
#define N_EDGES 160000
#define N_GRAPHS 64
#define HID 128
#define LAT 64

typedef _Float16 v16h __attribute__((ext_vector_type(16)));
typedef _Float16 v8h  __attribute__((ext_vector_type(8)));
typedef float    v8f  __attribute__((ext_vector_type(8)));

union AFrag { v16h v; v8h h2[2]; _Float16 e[16]; };
union CFrag { v8f v; float e[8]; };

// ---------------- LDS layout for edge kernel (dynamic shared) ----------------
// [0,131072)        W1 B-fragments, f16 swizzled: frag(c=K/32 0..7, t=N/16 0..15)
// [131072,196608)   W2 B-fragments: frag(c 0..7, t 0..7)
// [196608,197632)   b1 (256 f32)
// [197632,198656)   w1_edge column (256 f32)  (W1[:,256])
// [198656,199168)   b2 (128 f32)
// [199168,268800)   8 waves x 8704B: {16 dst i32, 16 src i32, 16 ea f32, pad, m1 staging 16x264 f16}
#define EDGE_LDS 268800
#define NODE_LDS (131584 + 2048 + 1024)

__device__ __forceinline__ float silu_f(float y) { return y / (1.f + __expf(-y)); }

// -------------------------------- embed --------------------------------------
__global__ void embed_kernel(const float* __restrict__ x,
                             const float* __restrict__ W, const float* __restrict__ b,
                             const float* __restrict__ lng, const float* __restrict__ lnb,
                             float* __restrict__ h) {
    __shared__ float red[HID];
    const int i = blockIdx.x, t = threadIdx.x;
    float x0 = x[i*3+0], x1 = x[i*3+1], x2 = x[i*3+2];
    float y = b[t] + x0*W[t*3] + x1*W[t*3+1] + x2*W[t*3+2];
    red[t] = y; __syncthreads();
    for (int s = 64; s > 0; s >>= 1) { if (t < s) red[t] += red[t+s]; __syncthreads(); }
    float mean = red[0] * (1.f/HID); __syncthreads();
    float d = y - mean; red[t] = d*d; __syncthreads();
    for (int s = 64; s > 0; s >>= 1) { if (t < s) red[t] += red[t+s]; __syncthreads(); }
    float var = red[0] * (1.f/HID);
    h[i*HID + t] = silu_f(d * rsqrtf(var + 1e-5f) * lng[t] + lnb[t]);
}

// -------------------------------- zero ---------------------------------------
__global__ void zero_kernel(float* __restrict__ p, int n) {
    int gid = blockIdx.x * 256 + threadIdx.x;
    if (gid < n) p[gid] = 0.f;
}

// ------------------------- edge MLP (WMMA core) -------------------------------
__global__ void __launch_bounds__(256, 1)
edge_mlp_kernel(const float* __restrict__ h,
                const float* __restrict__ edge_attr,
                const int*   __restrict__ ei,
                const float* __restrict__ msg1_W, const float* __restrict__ msg1_b,
                const float* __restrict__ msg2_W, const float* __restrict__ msg2_b,
                float* __restrict__ aggr, int layer) {
    extern __shared__ char smem[];
    float* b1s  = (float*)(smem + 196608);
    float* w1es = (float*)(smem + 197632);
    float* b2s  = (float*)(smem + 198656);
    char*  wsAll = smem + 199168;

    const int tid = threadIdx.x;
    // ---- stage W1 (K=0..255 part) as swizzled f16 B-fragments ----
    const float* W1g = msg1_W + (size_t)layer * 256 * 257;
    for (int idx = tid; idx < 256*256; idx += 256) {
        int n = idx >> 8, kk = idx & 255;
        float w = W1g[n*257 + kk];
        int c = kk >> 5, t = n >> 4, ko = kk & 31, g = (ko >> 3) & 1, hh = (ko & 7) | ((ko >> 4) << 3);
        int lane = (g << 4) | (n & 15);
        int addr = ((c*16 + t) << 10) + ((hh >> 3) << 9) + lane*16 + (hh & 7)*2;
        *(_Float16*)(smem + addr) = (_Float16)w;
    }
    if (tid < 256) { b1s[tid] = msg1_b[layer*256 + tid]; w1es[tid] = W1g[tid*257 + 256]; }
    const float* W2g = msg2_W + (size_t)layer * 128 * 256;
    for (int idx = tid; idx < 128*256; idx += 256) {
        int n = idx >> 8, kk = idx & 255;
        float w = W2g[n*256 + kk];
        int c = kk >> 5, t = n >> 4, ko = kk & 31, g = (ko >> 3) & 1, hh = (ko & 7) | ((ko >> 4) << 3);
        int lane = (g << 4) | (n & 15);
        int addr = 131072 + ((c*8 + t) << 10) + ((hh >> 3) << 9) + lane*16 + (hh & 7)*2;
        *(_Float16*)(smem + addr) = (_Float16)w;
    }
    if (tid < 128) b2s[tid] = msg2_b[layer*128 + tid];
    __syncthreads();

    const int wave = tid >> 5, lane = tid & 31;
    const int g = lane >> 4, lr = lane & 15;
    char* ws = wsAll + wave * 8704;
    int*   sdst = (int*)ws;
    int*   ssrc = (int*)(ws + 64);
    float* sea  = (float*)(ws + 128);
    _Float16* m1s = (_Float16*)(ws + 256);   // 16 rows x 264-half stride

    const int nTiles = N_EDGES / 16;
    for (int tile = blockIdx.x * 8 + wave; tile < nTiles; tile += gridDim.x * 8) {
        const int e0 = tile * 16;
        if (lane < 16) {
            ssrc[lane] = ei[e0 + lane];
            sdst[lane] = ei[N_EDGES + e0 + lane];
            sea[lane]  = edge_attr[e0 + lane];
        }
        // per-lane hoisted tile metadata (LDS -> registers, once per tile)
        const int nd = sdst[lr];          // gather node for K<128 (h[dst])
        const int ns = ssrc[lr];          // gather node for K>=128 (h[src])
        float ear[8];
        int   dstr[8];
#pragma unroll
        for (int j = 0; j < 8; ++j) {
            int row = j + (g << 3);
            ear[j]  = sea[row];
            dstr[j] = sdst[row];
        }
        // ---- build A fragments: z = [h[dst] | h[src]] as f16 ----
        AFrag a[8];
#pragma unroll
        for (int c = 0; c < 8; ++c) {
            int node = (c < 4) ? nd : ns;
            int f0 = ((c & 3) << 5) + (g << 3);
            const float4* hp4 = (const float4*)(h + node*HID + f0);
            float4 l0 = hp4[0], l1 = hp4[1], u0 = hp4[4], u1 = hp4[5];
            a[c].e[0]=(_Float16)l0.x; a[c].e[1]=(_Float16)l0.y; a[c].e[2]=(_Float16)l0.z; a[c].e[3]=(_Float16)l0.w;
            a[c].e[4]=(_Float16)l1.x; a[c].e[5]=(_Float16)l1.y; a[c].e[6]=(_Float16)l1.z; a[c].e[7]=(_Float16)l1.w;
            a[c].e[8]=(_Float16)u0.x; a[c].e[9]=(_Float16)u0.y; a[c].e[10]=(_Float16)u0.z; a[c].e[11]=(_Float16)u0.w;
            a[c].e[12]=(_Float16)u1.x; a[c].e[13]=(_Float16)u1.y; a[c].e[14]=(_Float16)u1.z; a[c].e[15]=(_Float16)u1.w;
        }
        // ---- GEMM1: 16x256, epilogue silu -> m1 staging (f16) ----
#pragma unroll 2
        for (int t = 0; t < 16; ++t) {
            CFrag acc;
#pragma unroll
            for (int j = 0; j < 8; ++j) acc.e[j] = 0.f;
#pragma unroll
            for (int c = 0; c < 8; ++c) {
                AFrag bf;
                const char* fb = smem + ((c*16 + t) << 10);
                bf.h2[0] = *(const v8h*)(fb + lane*16);
                bf.h2[1] = *(const v8h*)(fb + 512 + lane*16);
                acc.v = __builtin_amdgcn_wmma_f32_16x16x32_f16(
                    false, a[c].v, false, bf.v, (short)0, acc.v, false, false);
            }
            int n = (t << 4) + lr;
            float b1v = b1s[n], w1ev = w1es[n];
#pragma unroll
            for (int j = 0; j < 8; ++j) {
                int row = j + (g << 3);
                float y = acc.e[j] + b1v + ear[j] * w1ev;
                m1s[row*264 + n] = (_Float16)silu_f(y);
            }
        }
        // ---- GEMM2: 16x128, A fragments loaded once, epilogue scatter-add ----
        AFrag a2[8];
#pragma unroll
        for (int c = 0; c < 8; ++c) {
            const char* rb = (const char*)m1s + lr*528;
            int k0 = (c << 5) + (g << 3);
            a2[c].h2[0] = *(const v8h*)(rb + k0*2);
            a2[c].h2[1] = *(const v8h*)(rb + (k0 + 16)*2);
        }
#pragma unroll 2
        for (int t = 0; t < 8; ++t) {
            CFrag acc;
#pragma unroll
            for (int j = 0; j < 8; ++j) acc.e[j] = 0.f;
#pragma unroll
            for (int c = 0; c < 8; ++c) {
                AFrag bf;
                const char* fb = smem + 131072 + ((c*8 + t) << 10);
                bf.h2[0] = *(const v8h*)(fb + lane*16);
                bf.h2[1] = *(const v8h*)(fb + 512 + lane*16);
                acc.v = __builtin_amdgcn_wmma_f32_16x16x32_f16(
                    false, a2[c].v, false, bf.v, (short)0, acc.v, false, false);
            }
            int n2 = (t << 4) + lr;
            float b2v = b2s[n2];
#pragma unroll
            for (int j = 0; j < 8; ++j) {
                atomicAdd(aggr + dstr[j]*HID + n2, acc.e[j] + b2v);
            }
        }
    }
}

// --------------------------- gated node update + LN ---------------------------
__global__ void __launch_bounds__(256)
node_update_kernel(float* __restrict__ h, const float* __restrict__ aggr,
                   const float* __restrict__ gate_W, const float* __restrict__ gate_b,
                   const float* __restrict__ ln_g, const float* __restrict__ ln_b,
                   int layer) {
    extern __shared__ char smem[];
    float* gWs = (float*)smem;               // 128 x 257 (padded)
    float* zsh = (float*)(smem + 131584);    // 2 nodes x 256
    float* red = (float*)(smem + 133632);    // 256
    const int tid = threadIdx.x;
    const float* Wg = gate_W + (size_t)layer * 128 * 256;
    for (int idx = tid; idx < 128*256; idx += 256) {
        int n = idx >> 8, k = idx & 255;
        gWs[n*257 + k] = Wg[n*256 + k];
    }
    __syncthreads();
    const int nl = tid >> 7, n = tid & 127;
    const float gb = gate_b[layer*HID + n];
    const float lg = ln_g[layer*HID + n], lb = ln_b[layer*HID + n];
    for (int pair = blockIdx.x; pair < N_NODES/2; pair += gridDim.x) {
        int node = pair*2 + nl;
        zsh[nl*256 + n]       = h[node*HID + n];
        zsh[nl*256 + 128 + n] = aggr[node*HID + n];
        __syncthreads();
        float acc = gb;
        const float* wr = gWs + n*257;
        const float* zr = zsh + nl*256;
        for (int k = 0; k < 256; ++k) acc += wr[k] * zr[k];
        float gate = 1.f / (1.f + __expf(-acc));
        float hv = zr[n], av = zr[128 + n];
        float y = hv + gate*av + (1.f - gate)*hv;
        int base = nl*128;
        red[base + n] = y; __syncthreads();
        for (int s = 64; s > 0; s >>= 1) { if (n < s) red[base+n] += red[base+n+s]; __syncthreads(); }
        float mean = red[base] * (1.f/HID); __syncthreads();
        float d = y - mean; red[base + n] = d*d; __syncthreads();
        for (int s = 64; s > 0; s >>= 1) { if (n < s) red[base+n] += red[base+n+s]; __syncthreads(); }
        float var = red[base] * (1.f/HID); __syncthreads();
        h[node*HID + n] = d * rsqrtf(var + 1e-5f) * lg + lb;
        __syncthreads();
    }
}

// -------------------------------- pooling ------------------------------------
__global__ void pool_kernel(const float* __restrict__ h, const int* __restrict__ batch,
                            float* __restrict__ sums, float* __restrict__ counts) {
    int gid = blockIdx.x * 256 + threadIdx.x;
    if (gid >= N_NODES * HID) return;
    int i = gid >> 7, f = gid & 127;
    int g = batch[i];
    atomicAdd(sums + g*HID + f, h[gid]);
    if (f == 0) atomicAdd(counts + g, 1.0f);
}

// -------------------------------- projector ----------------------------------
__global__ void proj_kernel(const float* __restrict__ sums, const float* __restrict__ counts,
                            const float* __restrict__ p1W, const float* __restrict__ p1b,
                            const float* __restrict__ l1g, const float* __restrict__ l1b,
                            const float* __restrict__ p2W, const float* __restrict__ p2b,
                            const float* __restrict__ l2g, const float* __restrict__ l2b,
                            float* __restrict__ out) {
    __shared__ float xsh[HID], psh[HID], red[HID];
    const int t = threadIdx.x;
    for (int gi = 0; gi < N_GRAPHS; ++gi) {
        float c = counts[gi];
        float inv = 1.f / fmaxf(c, 1.f) + 1.f / (c + 1e-6f);
        float xg = sums[gi*HID + t] * inv;
        xsh[t] = xg;
        out[N_GRAPHS*LAT + gi*HID + t] = xg;
        __syncthreads();
        float y = p1b[t];
        for (int k = 0; k < HID; ++k) y += p1W[t*HID + k] * xsh[k];
        red[t] = y; __syncthreads();
        for (int s = 64; s > 0; s >>= 1) { if (t < s) red[t] += red[t+s]; __syncthreads(); }
        float mean = red[0] * (1.f/HID); __syncthreads();
        float d = y - mean; red[t] = d*d; __syncthreads();
        for (int s = 64; s > 0; s >>= 1) { if (t < s) red[t] += red[t+s]; __syncthreads(); }
        float var = red[0] * (1.f/HID); __syncthreads();
        psh[t] = silu_f(d * rsqrtf(var + 1e-5f) * l1g[t] + l1b[t]);
        __syncthreads();
        float y2 = 0.f;
        if (t < LAT) { y2 = p2b[t]; for (int k = 0; k < HID; ++k) y2 += p2W[t*HID + k] * psh[k]; }
        red[t] = (t < LAT) ? y2 : 0.f; __syncthreads();
        for (int s = 32; s > 0; s >>= 1) { if (t < s) red[t] += red[t+s]; __syncthreads(); }
        float mean2 = red[0] * (1.f/LAT); __syncthreads();
        float d2 = y2 - mean2;
        red[t] = (t < LAT) ? d2*d2 : 0.f; __syncthreads();
        for (int s = 32; s > 0; s >>= 1) { if (t < s) red[t] += red[t+s]; __syncthreads(); }
        float var2 = red[0] * (1.f/LAT); __syncthreads();
        if (t < LAT) out[gi*LAT + t] = d2 * rsqrtf(var2 + 1e-5f) * l2g[t] + l2b[t];
        __syncthreads();
    }
}

extern "C" void kernel_launch(void* const* d_in, const int* in_sizes, int n_in,
                              void* d_out, int out_size, void* d_ws, size_t ws_size,
                              hipStream_t stream) {
    const float* x         = (const float*)d_in[0];
    const float* edge_attr = (const float*)d_in[1];
    const int*   ei        = (const int*)d_in[2];
    const int*   batch     = (const int*)d_in[3];
    const float* emb_W = (const float*)d_in[4];
    const float* emb_b = (const float*)d_in[5];
    const float* emb_lg = (const float*)d_in[6];
    const float* emb_lb = (const float*)d_in[7];
    const float* msg1_W = (const float*)d_in[8];
    const float* msg1_b = (const float*)d_in[9];
    const float* msg2_W = (const float*)d_in[10];
    const float* msg2_b = (const float*)d_in[11];
    const float* gate_W = (const float*)d_in[12];
    const float* gate_b = (const float*)d_in[13];
    const float* ln_g = (const float*)d_in[14];
    const float* ln_b = (const float*)d_in[15];
    const float* p1W = (const float*)d_in[16];
    const float* p1b = (const float*)d_in[17];
    const float* l1g = (const float*)d_in[18];
    const float* l1b = (const float*)d_in[19];
    const float* p2W = (const float*)d_in[20];
    const float* p2b = (const float*)d_in[21];
    const float* l2g = (const float*)d_in[22];
    const float* l2b = (const float*)d_in[23];
    float* out = (float*)d_out;

    float* h      = (float*)d_ws;
    float* aggr   = (float*)((char*)d_ws + 5120000);
    float* sums   = (float*)((char*)d_ws + 10240000);
    float* counts = (float*)((char*)d_ws + 10272768);

    (void)hipFuncSetAttribute((const void*)edge_mlp_kernel,
                              hipFuncAttributeMaxDynamicSharedMemorySize, EDGE_LDS);
    (void)hipFuncSetAttribute((const void*)node_update_kernel,
                              hipFuncAttributeMaxDynamicSharedMemorySize, NODE_LDS);

    embed_kernel<<<N_NODES, HID, 0, stream>>>(x, emb_W, emb_b, emb_lg, emb_lb, h);
    for (int l = 0; l < 3; ++l) {
        zero_kernel<<<(N_NODES*HID + 255)/256, 256, 0, stream>>>(aggr, N_NODES*HID);
        edge_mlp_kernel<<<256, 256, EDGE_LDS, stream>>>(h, edge_attr, ei,
                                                        msg1_W, msg1_b, msg2_W, msg2_b, aggr, l);
        node_update_kernel<<<640, 256, NODE_LDS, stream>>>(h, aggr, gate_W, gate_b, ln_g, ln_b, l);
    }
    zero_kernel<<<(N_GRAPHS*HID + N_GRAPHS + 255)/256, 256, 0, stream>>>(sums, N_GRAPHS*HID + N_GRAPHS);
    pool_kernel<<<(N_NODES*HID + 255)/256, 256, 0, stream>>>(h, batch, sums, counts);
    proj_kernel<<<1, HID, 0, stream>>>(sums, counts, p1W, p1b, l1g, l1b, p2W, p2b, l2g, l2b, out);
}
